// BertBiAttention_52020643889861
// MI455X (gfx1250) — matrix operand order repaired and checked by reading, not compiled, not run on hardware
//
#include <hip/hip_runtime.h>
#include <stdint.h>

// ---------------------------------------------------------------------------
// BertBiAttention for MI455X (gfx1250).
//   B=8, S=1024, HID=512, H=8, D=64.
//   bf16 WMMA (v_wmma_f32_16x16x32_bf16) for all matmuls, fp32 softmax/probs.
//   Scores held entirely in VGPRs (wave32 -> 128 VGPRs for 16 v8f accums).
//   V stored transposed [B,H,D,S] so PV B-fragments are contiguous in global.
//   Q pre-scaled by 1/sqrt(D)=0.125 at projection time (exact in bf16).
//   Q tile staged to LDS with GLOBAL_LOAD_ASYNC_TO_LDS_B128 (ASYNCcnt path).
// ---------------------------------------------------------------------------

typedef __bf16 bf16_t;
typedef __attribute__((ext_vector_type(16))) __bf16 v16bf;
typedef __attribute__((ext_vector_type(8)))  __bf16 v8bf;
typedef __attribute__((ext_vector_type(8)))  float  v8f;

union V16U { v16bf v; v8bf h[2]; bf16_t e[16]; };
union V8BU { v8bf v; bf16_t e[8]; };

__device__ __forceinline__ v8f wmma_bf16(v16bf a, v16bf b, v8f c) {
  // (neg_a, A, neg_b, B, c_mod, C, reuse_a, reuse_b)
  return __builtin_amdgcn_wmma_f32_16x16x32_bf16(false, a, false, b,
                                                 (short)0, c, false, false);
}

// ---------------------------------------------------------------------------
// Kernel 1: six projections  C = (X @ W^T + b) * oscale  -> bf16.
//   Q (z=0,3): [B,H,S,D], scaled by 0.125 (exact: power-of-two in bf16).
//   K (z=1,4): [B,H,S,D].
//   V (z=2,5): [B,H,D,S] (transposed).
//   Block: 64(M) x 64(N), 8 waves, wave = 1 row-tile x 2 col-tiles.
// ---------------------------------------------------------------------------

struct ProjArgs {
  const float* X[6];
  const float* W[6];
  const float* bias[6];
  bf16_t*      out[6];
};

__global__ __launch_bounds__(256) void bertbi_proj_kernel(ProjArgs args) {
  const int z = blockIdx.z;
  const float* __restrict__ X    = args.X[z];
  const float* __restrict__ W    = args.W[z];
  const float* __restrict__ bias = args.bias[z];
  bf16_t*      __restrict__ out  = args.out[z];
  const bool  transposeV = (z == 2 || z == 5);
  const float oscale     = (z == 0 || z == 3) ? 0.125f : 1.0f;

  __shared__ __align__(16) bf16_t As[64][40];  // 32 + pad (80B row stride)
  __shared__ __align__(16) bf16_t Bs[64][40];

  const int tid  = threadIdx.x;
  const int lane = tid & 31;
  const int wave = tid >> 5;
  const int m0 = blockIdx.y * 64;
  const int n0 = blockIdx.x * 64;
  const int rt  = wave & 3;
  const int ct0 = (wave >> 2) * 2;
  const int l15   = lane & 15;
  const int khalf = lane >> 4;

  v8f acc[2] = {};

  for (int k0 = 0; k0 < 512; k0 += 32) {
    __syncthreads();
    {   // cooperative tile load + f32->bf16
      const int r = tid >> 2;
      const int c = (tid & 3) * 8;
      const float* xp = X + (size_t)(m0 + r) * 512 + k0 + c;
      const float* wp = W + (size_t)(n0 + r) * 512 + k0 + c;
#pragma unroll
      for (int i = 0; i < 8; ++i) {
        As[r][c + i] = (bf16_t)xp[i];
        Bs[r][c + i] = (bf16_t)wp[i];
      }
    }
    __syncthreads();

    V16U a, b0, b1;
    // A frag 16x32: row=l15, k=(i<8? i : i+8)+8*khalf
    const bf16_t* arow = &As[16 * rt + l15][0];
    a.h[0] = *(const v8bf*)(arow + 8 * khalf);
    a.h[1] = *(const v8bf*)(arow + 16 + 8 * khalf);
    // B frag 32x16: col=l15, k=(i<8? i : i+8)+8*khalf
    const bf16_t* brow0 = &Bs[16 * ct0 + l15][0];
    const bf16_t* brow1 = &Bs[16 * (ct0 + 1) + l15][0];
    b0.h[0] = *(const v8bf*)(brow0 + 8 * khalf);
    b0.h[1] = *(const v8bf*)(brow0 + 16 + 8 * khalf);
    b1.h[0] = *(const v8bf*)(brow1 + 8 * khalf);
    b1.h[1] = *(const v8bf*)(brow1 + 16 + 8 * khalf);

    acc[0] = wmma_bf16(a.v, b0.v, acc[0]);
    acc[1] = wmma_bf16(a.v, b1.v, acc[1]);
  }

  // epilogue: + bias, * oscale, store bf16
#pragma unroll
  for (int j = 0; j < 2; ++j) {
    const int n_local = 16 * (ct0 + j) + l15;   // C layout: N = lane&15
    const int gn = n0 + n_local;
    const int hh = gn >> 6, dd = gn & 63;
    const float bv = bias[gn];
    if (transposeV) {
      // [B,H,D,S]: the 8 accum rows are 8 consecutive s -> one 16B store
      V8BU o;
#pragma unroll
      for (int r = 0; r < 8; ++r) o.e[r] = (bf16_t)(acc[j][r] + bv);
      const int gm0 = m0 + 16 * rt + 8 * khalf;
      const int bb = gm0 >> 10, ss = gm0 & 1023;
      *(v8bf*)&out[((((size_t)bb * 8 + hh) * 64 + dd) << 10) + ss] = o.v;
    } else {
      // [B,H,S,D]
#pragma unroll
      for (int r = 0; r < 8; ++r) {
        const int gm = m0 + 16 * rt + r + 8 * khalf;
        const int bb = gm >> 10, ss = gm & 1023;
        out[((((size_t)bb * 8 + hh) * 1024 + ss) << 6) + dd] =
            (bf16_t)((acc[j][r] + bv) * oscale);
      }
    }
  }
}

// ---------------------------------------------------------------------------
// Kernel 2: one (b,h), 32 query rows per block, 8 waves.
//   Scores live in VGPRs: wave (rt=w&1, kg=w>>1) holds rows [16rt,16rt+16)
//   x keys [kg*256, kg*256+256) in 16 v8f accumulators.
//   Softmax: in-register + 16-lane shfl_xor + 1KB LDS cross-wave reduce.
//   probs: fp32 non-temporal stores straight from registers.
//   P staged once as bf16 in LDS; PV reads V from the transposed [B,H,D,S]
//   global layout -> 2x16B contiguous loads per B-fragment, no barriers.
// ---------------------------------------------------------------------------

#define PB_STRIDE 1048   // bf16 elements per P row (1024 + pad), 2096B, 16B mult

__global__ __launch_bounds__(256) void bertbi_attn_kernel(
    const bf16_t* __restrict__ Q, const bf16_t* __restrict__ Km,
    const bf16_t* __restrict__ Vt, const float* __restrict__ mask,
    float* __restrict__ probs, float* __restrict__ ctx) {
  extern __shared__ char smem_raw[];
  bf16_t* Qs   = (bf16_t*)smem_raw;                       // 32x72   (4608 B)
  bf16_t* Pbf  = (bf16_t*)(smem_raw + 4608);              // 32x1048 (67072 B)
  float*  redm = (float*)(smem_raw + 4608 + 67072);       // 32x4 max
  float*  reds = redm + 32 * 4;                           // 32x4 sum

  const int tid  = threadIdx.x;
  const int lane = tid & 31;
  const int wave = tid >> 5;
  const int l15   = lane & 15;
  const int khalf = lane >> 4;
  const int bh = blockIdx.y;
  const int b  = bh >> 3;
  const int hh = bh & 7;
  const int q0 = blockIdx.x * 32;
  const size_t baseQ = ((size_t)bh * 1024 + q0) * 64;
  const size_t baseK = (size_t)bh * 1024 * 64;   // K: [B,H,S,D]
  const size_t baseV = (size_t)bh * 64 * 1024;   // V: [B,H,D,S]

  // ---- stage 32x64 Q tile in LDS via async global->LDS DMA (ASYNCcnt) ----
  {
    const int r = tid >> 3, seg = (tid & 7) * 8;
    // flat LDS pointers carry the LDS byte offset in addr[31:0] (ISA 10.2)
    const unsigned ldsoff = (unsigned)(uintptr_t)&Qs[r * 72 + seg];
    const uint64_t gaddr =
        (uint64_t)(uintptr_t)(Q + baseQ + (size_t)r * 64 + seg);
    asm volatile("global_load_async_to_lds_b128 %0, %1, off"
                 :: "v"(ldsoff), "v"(gaddr)
                 : "memory");
    asm volatile("s_wait_asynccnt 0x0" ::: "memory");
  }
  __syncthreads();

  const int rt = wave & 1;     // q row tile
  const int kg = wave >> 1;    // key quarter 0..3 (keys kg*256..+255)

  // ---- phase 1: scores (held in registers; Q already carries 1/sqrt(D)) ----
  V16U a0, a1;
  {
    const bf16_t* qrow = &Qs[(16 * rt + l15) * 72];
    a0.h[0] = *(const v8bf*)(qrow + 8 * khalf);
    a0.h[1] = *(const v8bf*)(qrow + 16 + 8 * khalf);
    a1.h[0] = *(const v8bf*)(qrow + 32 + 8 * khalf);
    a1.h[1] = *(const v8bf*)(qrow + 48 + 8 * khalf);
  }

  v8f acc[16];
#pragma unroll
  for (int j = 0; j < 16; ++j) {
    const int kc = kg * 16 + j;
    const bf16_t* krow = Km + baseK + (size_t)(kc * 16 + l15) * 64;
    V16U b0, b1;                     // B[k=d][n=key] = K[key][d], contiguous d
    b0.h[0] = *(const v8bf*)(krow + 8 * khalf);
    b0.h[1] = *(const v8bf*)(krow + 16 + 8 * khalf);
    b1.h[0] = *(const v8bf*)(krow + 32 + 8 * khalf);
    b1.h[1] = *(const v8bf*)(krow + 48 + 8 * khalf);
    v8f t = {};
    t = wmma_bf16(a0.v, b0.v, t);
    acc[j] = wmma_bf16(a1.v, b1.v, t);
  }

  // prefetch the V row this wave will consume in phase 3
  const int dd_pf = 16 * (wave >> 1) + l15;
  __builtin_prefetch(Vt + baseV + (size_t)dd_pf * 1024, 0, 1);

  // ---- phase 2: * mask, softmax, write probs ----
  const float* mrow = mask + ((size_t)b * 1024 + q0) * 1024;
#pragma unroll
  for (int j = 0; j < 16; ++j) {
    const int kk = (kg * 16 + j) * 16 + l15;
#pragma unroll
    for (int r = 0; r < 8; ++r) {
      const int qloc = 16 * rt + r + 8 * khalf;
      acc[j][r] *= mrow[(size_t)qloc * 1024 + kk];
    }
  }

  // partial row max over this wave's 256 keys
#pragma unroll
  for (int r = 0; r < 8; ++r) {
    float m = acc[0][r];
#pragma unroll
    for (int j = 1; j < 16; ++j) m = fmaxf(m, acc[j][r]);
#pragma unroll
    for (int off = 8; off > 0; off >>= 1) m = fmaxf(m, __shfl_xor(m, off));
    if (l15 == 0) redm[(16 * rt + r + 8 * khalf) * 4 + kg] = m;
  }
  __syncthreads();

  float fmx[8];
#pragma unroll
  for (int r = 0; r < 8; ++r) {
    const int qloc = 16 * rt + r + 8 * khalf;
    fmx[r] = fmaxf(fmaxf(redm[qloc * 4 + 0], redm[qloc * 4 + 1]),
                   fmaxf(redm[qloc * 4 + 2], redm[qloc * 4 + 3]));
  }

  float rsum[8];
#pragma unroll
  for (int r = 0; r < 8; ++r) rsum[r] = 0.f;
#pragma unroll
  for (int j = 0; j < 16; ++j) {
#pragma unroll
    for (int r = 0; r < 8; ++r) {
      const float e = __expf(acc[j][r] - fmx[r]);
      acc[j][r] = e;
      rsum[r] += e;
    }
  }
#pragma unroll
  for (int r = 0; r < 8; ++r) {
#pragma unroll
    for (int off = 8; off > 0; off >>= 1) rsum[r] += __shfl_xor(rsum[r], off);
    if (l15 == 0) reds[(16 * rt + r + 8 * khalf) * 4 + kg] = rsum[r];
  }
  __syncthreads();

  float inv[8];
#pragma unroll
  for (int r = 0; r < 8; ++r) {
    const int qloc = 16 * rt + r + 8 * khalf;
    inv[r] = 1.0f / (reds[qloc * 4 + 0] + reds[qloc * 4 + 1] +
                     reds[qloc * 4 + 2] + reds[qloc * 4 + 3]);
  }

  // normalize: probs (fp32, streaming NT stores) + P bf16 into LDS
#pragma unroll
  for (int j = 0; j < 16; ++j) {
    const int kk = (kg * 16 + j) * 16 + l15;
#pragma unroll
    for (int r = 0; r < 8; ++r) {
      const int qloc = 16 * rt + r + 8 * khalf;
      const float p = acc[j][r] * inv[r];
      __builtin_nontemporal_store(
          p, &probs[((size_t)bh * 1024 + q0 + qloc) * 1024 + kk]);
      Pbf[qloc * PB_STRIDE + kk] = (bf16_t)p;
    }
  }
  __syncthreads();

  // ---- phase 3: ctx = P @ V  (V transposed in global: contiguous B-frags) ----
  {
    const int crt = wave & 1;     // q row tile
    const int cct = wave >> 1;    // d col tile 0..3
    const int dd = 16 * cct + l15;
    const bf16_t* arow = &Pbf[(16 * crt + l15) * PB_STRIDE];
    const bf16_t* vrow = Vt + baseV + (size_t)dd * 1024;
    v8f o = {};
#pragma unroll 4
    for (int kc = 0; kc < 32; ++kc) {
      V16U a, bv;
      a.h[0] = *(const v8bf*)(arow + kc * 32 + 8 * khalf);
      a.h[1] = *(const v8bf*)(arow + kc * 32 + 16 + 8 * khalf);
      bv.h[0] = *(const v8bf*)(vrow + kc * 32 + 8 * khalf);
      bv.h[1] = *(const v8bf*)(vrow + kc * 32 + 16 + 8 * khalf);
      o = wmma_bf16(a.v, bv.v, o);
    }
#pragma unroll
    for (int r = 0; r < 8; ++r) {
      const int qloc = 16 * crt + r + 8 * khalf;
      __builtin_nontemporal_store(
          o[r], &ctx[((size_t)b * 1024 + q0 + qloc) * 512 + hh * 64 + dd]);
    }
  }
}

// ---------------------------------------------------------------------------
// Host launcher
// ---------------------------------------------------------------------------
extern "C" void kernel_launch(void* const* d_in, const int* in_sizes, int n_in,
                              void* d_out, int out_size, void* d_ws,
                              size_t ws_size, hipStream_t stream) {
  (void)in_sizes; (void)n_in; (void)out_size; (void)ws_size;

  const float* X1    = (const float*)d_in[0];
  const float* mask1 = (const float*)d_in[1];
  const float* X2    = (const float*)d_in[2];
  const float* mask2 = (const float*)d_in[3];

  // workspace: bf16 Q,K ([B,H,S,D]) and V ([B,H,D,S]); 8 MB each, 48 MB total
  const size_t SZ = (size_t)8 * 8 * 1024 * 64;
  bf16_t* Q1 = (bf16_t*)d_ws;   // pre-scaled by 0.125
  bf16_t* K1 = Q1 + SZ;
  bf16_t* V1 = K1 + SZ;         // transposed
  bf16_t* Q2 = V1 + SZ;         // pre-scaled by 0.125
  bf16_t* K2 = Q2 + SZ;
  bf16_t* V2 = K2 + SZ;         // transposed

  ProjArgs pa;
  pa.X[0] = X1; pa.X[1] = X1; pa.X[2] = X1;
  pa.X[3] = X2; pa.X[4] = X2; pa.X[5] = X2;
  pa.W[0] = (const float*)d_in[4];  pa.bias[0] = (const float*)d_in[5];   // Wq1
  pa.W[1] = (const float*)d_in[6];  pa.bias[1] = (const float*)d_in[7];   // Wk1
  pa.W[2] = (const float*)d_in[8];  pa.bias[2] = (const float*)d_in[9];   // Wv1
  pa.W[3] = (const float*)d_in[10]; pa.bias[3] = (const float*)d_in[11];  // Wq2
  pa.W[4] = (const float*)d_in[12]; pa.bias[4] = (const float*)d_in[13];  // Wk2
  pa.W[5] = (const float*)d_in[14]; pa.bias[5] = (const float*)d_in[15];  // Wv2
  pa.out[0] = Q1; pa.out[1] = K1; pa.out[2] = V1;
  pa.out[3] = Q2; pa.out[4] = K2; pa.out[5] = V2;

  bertbi_proj_kernel<<<dim3(8, 128, 6), dim3(256), 0, stream>>>(pa);

  float* out    = (float*)d_out;
  float* ctx1   = out;                                  // [8,1024,512]
  float* ctx2   = ctx1 + (size_t)8 * 1024 * 512;        // [8,1024,512]
  float* probs1 = ctx2 + (size_t)8 * 1024 * 512;        // [8,8,1024,1024]
  float* probs2 = probs1 + (size_t)8 * 8 * 1024 * 1024; // [8,8,1024,1024]

  const size_t smem = 4608 + 67072 + 2 * 32 * 4 * sizeof(float); // 72704 B
  (void)hipFuncSetAttribute((const void*)bertbi_attn_kernel,
                            hipFuncAttributeMaxDynamicSharedMemorySize,
                            (int)smem);

  // stream 1: q2 over (k1, v1) -> ctx1/probs1; stream 2: q1 over (k2, v2)
  bertbi_attn_kernel<<<dim3(32, 64), dim3(256), smem, stream>>>(
      Q2, K1, V1, mask1, probs1, ctx1);
  bertbi_attn_kernel<<<dim3(32, 64), dim3(256), smem, stream>>>(
      Q1, K2, V2, mask2, probs2, ctx2);
}
